// MultiHead_89575837926046
// MI455X (gfx1250) — matrix operand order repaired and checked by reading, not compiled, not run on hardware
//
#include <hip/hip_runtime.h>

// Multi-head attention (B=2,S=2048,D=1024,H=16,HD=64) on gfx1250.
// All matmuls via v_wmma_f32_16x16x32_f16; flash-attention K/V staging via
// GLOBAL_LOAD_ASYNC_TO_LDS_B128 (ASYNCcnt) with double-buffered LDS tiles.
// Pipeline:
//   1) Qp = q@Wk+bk -> f16 [B,H,S,HD]   (reference bug reproduced: q uses Wk)
//   2) Kp = k@Wk+bk -> f16 [B,H,S,HD]
//   3) Vt = v@Wv+bv -> f16 [B,H,HD,S]   (transposed: P*V B-frags contiguous)
//   4) flash attention (online softmax, causal) -> Ow f16 [B,S,H*HD]
//   5) out = Ow@Wo+bo -> f32 [B,S,D]
// Workspace: 4 * 8MB f16 = 32 MB in d_ws.

typedef _Float16 f16;
typedef _Float16 v16h __attribute__((ext_vector_type(16)));
typedef _Float16 v8h  __attribute__((ext_vector_type(8)));
typedef float    v8f  __attribute__((ext_vector_type(8)));

#define DEVINL __device__ __forceinline__

constexpr int BB   = 2;
constexpr int SS   = 2048;
constexpr int DD   = 1024;
constexpr int HH   = 16;
constexpr int HDIM = 64;
constexpr int MROWS = BB * SS;   // 4096

DEVINL v8f zero8() {
  v8f z;
#pragma unroll
  for (int i = 0; i < 8; ++i) z[i] = 0.0f;
  return z;
}

// Async copy 16B from global to LDS (ASYNCcnt-tracked, no VGPR staging).
DEVINL void async_b128(unsigned lds_off, const void* gptr) {
  asm volatile("global_load_async_to_lds_b128 %0, %1, off"
               :: "v"(lds_off), "v"(gptr) : "memory");
}
DEVINL void wait_async() {
  asm volatile("s_wait_asynccnt 0" ::: "memory");
}
DEVINL unsigned lds_off(const void* p) {
  return (unsigned)(unsigned long long)p;   // low 32 bits of LDS generic addr
}

// A-fragment (16x32 f16, MxK): lane m = lane&15, hi = lane>>4.
// halves j in [0,8): K = hi*8 + j ; j in [8,16): K = 16 + hi*8 + (j-8).
DEVINL v16h load_a_frag(const f16* p, int ld) {
  const int lane = threadIdx.x & 31;
  const int m = lane & 15, hi = lane >> 4;
  const f16* base = p + m * ld + hi * 8;
  v8h x = *(const v8h*)(base);
  v8h y = *(const v8h*)(base + 16);
  v16h r;
#pragma unroll
  for (int j = 0; j < 8; ++j) { r[j] = x[j]; r[j + 8] = y[j]; }
  return r;
}

// B-fragment (32x16 f16, KxN) from a TRANSPOSED LDS tile [N][K]:
// lane n = lane&15, j<8: K = hi*8 + j ; j>=8: K = 16 + hi*8 + (j-8).
DEVINL v16h load_bT_frag(const f16* p, int ld) {
  const int lane = threadIdx.x & 31;
  const int n = lane & 15, hi = lane >> 4;
  const f16* base = p + n * ld + hi * 8;
  v8h x = *(const v8h*)(base);
  v8h y = *(const v8h*)(base + 16);
  v16h r;
#pragma unroll
  for (int j = 0; j < 8; ++j) { r[j] = x[j]; r[j + 8] = y[j]; }
  return r;
}

DEVINL v8f wmma_f16(v16h a, v16h b, v8f c) {
  return __builtin_amdgcn_wmma_f32_16x16x32_f16(
      false, a, false, b, (short)0, c, false, false);
}

// ---------------------------------------------------------------------------
// WMMA GEMM: out = A[M,K] * W[K,N] + bias[N].  AMODE: 0=f32 A, 1=f16 A.
// OMODE: 0=f32 [M,N]; 1=f16 [B,H,S,HD]; 2=f16 [B,H,HD,S].
// Block tile 128x128, 256 threads (8 waves, 4Mx2N), wave tile 32x64, BK=32.
// Double-buffered LDS: global loads for tile t+1 in flight during tile t WMMAs.
// ---------------------------------------------------------------------------
template <int AMODE, int OMODE>
__global__ void __launch_bounds__(256)
gemm128(const void* __restrict__ A_, const float* __restrict__ W,
        const float* __restrict__ bias, void* __restrict__ out_,
        int M, int N, int K) {
  __shared__ f16 As[2][128 * 40];   // [m][k], ld=40
  __shared__ f16 Wt[2][128 * 40];   // [n][k] (transposed W tile)

  const int tid  = threadIdx.x;
  const int lane = tid & 31, wid = tid >> 5;
  const int wm = wid & 3, wn = wid >> 2;
  const int m0 = blockIdx.y * 128, n0 = blockIdx.x * 128;

  float aR[16], wR[16];
  auto loadRegs = [&](int kt) {
#pragma unroll
    for (int i = 0; i < 16; ++i) {
      const int idx = tid + i * 256;
      const int m = idx >> 5, k = idx & 31;
      if (AMODE == 0)
        aR[i] = ((const float*)A_)[(size_t)(m0 + m) * K + kt + k];
      else
        aR[i] = (float)((const f16*)A_)[(size_t)(m0 + m) * K + kt + k];
    }
#pragma unroll
    for (int i = 0; i < 16; ++i) {
      const int idx = tid + i * 256;
      const int k = idx >> 7, n = idx & 127;
      wR[i] = W[(size_t)(kt + k) * N + n0 + n];
    }
  };
  auto storeRegs = [&](int buf) {
#pragma unroll
    for (int i = 0; i < 16; ++i) {
      const int idx = tid + i * 256;
      const int m = idx >> 5, k = idx & 31;
      As[buf][m * 40 + k] = (f16)aR[i];
    }
#pragma unroll
    for (int i = 0; i < 16; ++i) {
      const int idx = tid + i * 256;
      const int k = idx >> 7, n = idx & 127;
      Wt[buf][n * 40 + k] = (f16)wR[i];
    }
  };

  v8f acc[2][4];
#pragma unroll
  for (int i = 0; i < 2; ++i)
#pragma unroll
    for (int j = 0; j < 4; ++j) acc[i][j] = zero8();

  const int NT = K / 32;
  loadRegs(0);
  storeRegs(0);
  for (int t = 0; t < NT; ++t) {
    __syncthreads();                       // buf[t&1] populated; prev reads done
    if (t + 1 < NT) loadRegs((t + 1) * 32);  // global loads overlap WMMAs below
    const f16* Ab = &As[t & 1][0];
    const f16* Wb = &Wt[t & 1][0];
    const v16h a0 = load_a_frag(&Ab[(wm * 32) * 40], 40);
    const v16h a1 = load_a_frag(&Ab[(wm * 32 + 16) * 40], 40);
#pragma unroll
    for (int nj = 0; nj < 4; ++nj) {
      const v16h bb = load_bT_frag(&Wb[(wn * 64 + nj * 16) * 40], 40);
      acc[0][nj] = wmma_f16(a0, bb, acc[0][nj]);
      acc[1][nj] = wmma_f16(a1, bb, acc[1][nj]);
    }
    if (t + 1 < NT) storeRegs((t + 1) & 1);  // other buffer: no barrier needed
  }

  // Epilogue: bias + layout-specific store.
  const int hi = lane >> 4, nl = lane & 15;
#pragma unroll
  for (int mi = 0; mi < 2; ++mi) {
#pragma unroll
    for (int nj = 0; nj < 4; ++nj) {
      const int col = n0 + wn * 64 + nj * 16 + nl;
      const float bv = bias[col];
#pragma unroll
      for (int r = 0; r < 8; ++r) {
        const int row = m0 + wm * 32 + mi * 16 + r + 8 * hi;
        const float v = acc[mi][nj][r] + bv;
        if (OMODE == 0) {
          ((float*)out_)[(size_t)row * N + col] = v;
        } else {
          const int b = row / SS, s = row % SS;
          const int h = col / HDIM, d = col % HDIM;
          size_t o;
          if (OMODE == 1) o = ((size_t)(b * HH + h) * SS + s) * HDIM + d;
          else            o = ((size_t)(b * HH + h) * HDIM + d) * SS + s;
          ((f16*)out_)[o] = (f16)v;
        }
      }
    }
  }
}

// ---------------------------------------------------------------------------
// Flash attention (causal). One block = (b, h, 128 query rows); 4 waves,
// each wave owns 32 query rows (two 16-row subtiles). KV blocks of 64 with
// async-to-LDS double buffering; online softmax per row.
// ---------------------------------------------------------------------------
__global__ void __launch_bounds__(128)
flash_attn(const f16* __restrict__ Qp, const f16* __restrict__ Kp,
           const f16* __restrict__ Vt, f16* __restrict__ O) {
  __shared__ f16 Qs[128 * 72];       // [q][d]
  __shared__ f16 Ksb[2][64 * 72];    // [kv][d]   (B^T layout for S = Q*K^T)
  __shared__ f16 Vsb[2][64 * 72];    // [d][kv]   (B^T layout for O = P*V)
  __shared__ f16 Ps[4][32 * 72];     // per-wave P tile [q][kv]

  const int tid = threadIdx.x, lane = tid & 31, wid = tid >> 5;
  const int hi = lane >> 4, nl = lane & 15;
  const int q0 = blockIdx.x * 128;
  const int h = blockIdx.y, b = blockIdx.z;

  const f16* Qbh = Qp + (size_t)(b * HH + h) * SS * HDIM;
  const f16* Kbh = Kp + (size_t)(b * HH + h) * SS * HDIM;
  const f16* Vbh = Vt + (size_t)(b * HH + h) * HDIM * SS;

  // Async-stage Q tile (128x64): 1024 16B chunks, 8 per thread.
#pragma unroll
  for (int i = 0; i < 8; ++i) {
    const int ch = tid + i * 128;
    const int r = ch >> 3, cc = (ch & 7) * 8;
    async_b128(lds_off(&Qs[r * 72 + cc]),
               &Qbh[(size_t)(q0 + r) * HDIM + cc]);
  }
  // Async-stage KV tile 0 (64x64 each): 512 chunks, 4 per thread.
  auto stageKV = [&](int buf, int j0) {
#pragma unroll
    for (int i = 0; i < 4; ++i) {
      const int ch = tid + i * 128;
      const int r = ch >> 3, cc = (ch & 7) * 8;
      async_b128(lds_off(&Ksb[buf][r * 72 + cc]),
                 &Kbh[(size_t)(j0 + r) * HDIM + cc]);
      async_b128(lds_off(&Vsb[buf][r * 72 + cc]),
                 &Vbh[(size_t)r * SS + j0 + cc]);
    }
  };
  stageKV(0, 0);
  wait_async();
  __syncthreads();

  v16h qa[2][2];
#pragma unroll
  for (int mi = 0; mi < 2; ++mi) {
    qa[mi][0] = load_a_frag(&Qs[(wid * 32 + mi * 16) * 72 + 0], 72);
    qa[mi][1] = load_a_frag(&Qs[(wid * 32 + mi * 16) * 72 + 32], 72);
  }

  float mr[2][8], lr[2][8];
  v8f accO[2][4];
#pragma unroll
  for (int mi = 0; mi < 2; ++mi) {
#pragma unroll
    for (int r = 0; r < 8; ++r) { mr[mi][r] = -1e30f; lr[mi][r] = 0.0f; }
#pragma unroll
    for (int nd = 0; nd < 4; ++nd) accO[mi][nd] = zero8();
  }

  const float sc = 1.0f / (8.0f + 1e-8f);   // 1/(sqrt(HD)+1e-8)

  const int nkb = 2 * blockIdx.x + 2;       // causal KV-block bound
  for (int jb = 0; jb < nkb; ++jb) {
    const int j0 = jb * 64;
    const int cur = jb & 1;
    if (jb + 1 < nkb) stageKV(1 - cur, j0 + 64);  // overlap with compute below
    const f16* Ks = &Ksb[cur][0];
    const f16* Vs = &Vsb[cur][0];

#pragma unroll
    for (int mi = 0; mi < 2; ++mi) {
      // S = Q*K^T (contract HD=64 in two 32-wide K-steps).
      v8f sacc[4];
#pragma unroll
      for (int nj = 0; nj < 4; ++nj) {
        v8f a = zero8();
        const v16h b0 = load_bT_frag(&Ks[(nj * 16) * 72 + 0], 72);
        const v16h b1 = load_bT_frag(&Ks[(nj * 16) * 72 + 32], 72);
        a = wmma_f16(qa[mi][0], b0, a);
        a = wmma_f16(qa[mi][1], b1, a);
        sacc[nj] = a;
      }

      // Scale + causal mask (reference adds -1e9 after scaling).
      float sv[4][8];
#pragma unroll
      for (int nj = 0; nj < 4; ++nj)
#pragma unroll
        for (int r = 0; r < 8; ++r) {
          const int row = q0 + wid * 32 + mi * 16 + r + 8 * hi;
          const int col = j0 + nj * 16 + nl;
          float x = sacc[nj][r] * sc;
          if (col > row) x += -1e9f;
          sv[nj][r] = x;
        }

      // Online softmax; row stats reduce across the 16-lane half-wave.
#pragma unroll
      for (int r = 0; r < 8; ++r) {
        float tmax = sv[0][r];
#pragma unroll
        for (int nj = 1; nj < 4; ++nj) tmax = fmaxf(tmax, sv[nj][r]);
#pragma unroll
        for (int m = 8; m >= 1; m >>= 1) tmax = fmaxf(tmax, __shfl_xor(tmax, m));
        const float mnew = fmaxf(mr[mi][r], tmax);
        const float corr = __expf(mr[mi][r] - mnew);
        float rsum = 0.0f;
#pragma unroll
        for (int nj = 0; nj < 4; ++nj) {
          const float p = __expf(sv[nj][r] - mnew);
          sv[nj][r] = p;
          rsum += p;
        }
#pragma unroll
        for (int m = 8; m >= 1; m >>= 1) rsum += __shfl_xor(rsum, m);
        lr[mi][r] = lr[mi][r] * corr + rsum;
        mr[mi][r] = mnew;
#pragma unroll
        for (int nd = 0; nd < 4; ++nd) accO[mi][nd][r] *= corr;
      }

      // C-layout -> A-layout for P via per-wave LDS round trip
      // (same-wave DS ops execute in order; no barrier needed).
      f16* Pw = &Ps[wid][(mi * 16) * 72];
#pragma unroll
      for (int nj = 0; nj < 4; ++nj)
#pragma unroll
        for (int r = 0; r < 8; ++r)
          Pw[(r + 8 * hi) * 72 + nj * 16 + nl] = (f16)sv[nj][r];

      const v16h pa0 = load_a_frag(Pw + 0, 72);
      const v16h pa1 = load_a_frag(Pw + 32, 72);
#pragma unroll
      for (int nd = 0; nd < 4; ++nd) {
        const v16h b0 = load_bT_frag(&Vs[(nd * 16) * 72 + 0], 72);
        const v16h b1 = load_bT_frag(&Vs[(nd * 16) * 72 + 32], 72);
        accO[mi][nd] = wmma_f16(pa0, b0, accO[mi][nd]);
        accO[mi][nd] = wmma_f16(pa1, b1, accO[mi][nd]);
      }
    }

    if (jb + 1 < nkb) {
      wait_async();      // next KV tile landed while we computed this one
      __syncthreads();   // all waves done reading buf[cur] before next overwrite
    }
  }

  // Normalize and store O as [B,S,H,HD] f16 (merged-head row-major).
#pragma unroll
  for (int mi = 0; mi < 2; ++mi)
#pragma unroll
    for (int nd = 0; nd < 4; ++nd) {
      const int d = nd * 16 + nl;
#pragma unroll
      for (int r = 0; r < 8; ++r) {
        const int srow = q0 + wid * 32 + mi * 16 + r + 8 * hi;
        const float v = accO[mi][nd][r] / lr[mi][r];
        O[(((size_t)b * SS + srow) * HH + h) * HDIM + d] = (f16)v;
      }
    }
}

extern "C" void kernel_launch(void* const* d_in, const int* in_sizes, int n_in,
                              void* d_out, int out_size, void* d_ws, size_t ws_size,
                              hipStream_t stream) {
  (void)in_sizes; (void)n_in; (void)out_size; (void)ws_size;

  const float* q  = (const float*)d_in[0];
  const float* k  = (const float*)d_in[1];
  const float* v  = (const float*)d_in[2];
  // d_in[3] = mask : causal, computed analytically in-kernel.
  const float* Wk = (const float*)d_in[4];
  const float* bk = (const float*)d_in[5];
  const float* Wv = (const float*)d_in[6];
  const float* bv = (const float*)d_in[7];
  const float* Wo = (const float*)d_in[8];
  const float* bo = (const float*)d_in[9];
  float* out = (float*)d_out;

  const size_t HSZ = (size_t)BB * HH * SS * HDIM;  // 4Mi halves = 8 MB
  f16* Qp = (f16*)d_ws;
  f16* Kp = Qp + HSZ;
  f16* Vtp = Kp + HSZ;
  f16* Ow = Vtp + HSZ;   // requires ws_size >= 32 MB

  dim3 ggrid(DD / 128, MROWS / 128);  // (8, 32)
  gemm128<0, 1><<<ggrid, 256, 0, stream>>>(q, Wk, bk, Qp, MROWS, DD, DD);
  gemm128<0, 1><<<ggrid, 256, 0, stream>>>(k, Wk, bk, Kp, MROWS, DD, DD);
  gemm128<0, 2><<<ggrid, 256, 0, stream>>>(v, Wv, bv, Vtp, MROWS, DD, DD);

  dim3 fgrid(SS / 128, HH, BB);       // (16, 16, 2)
  flash_attn<<<fgrid, 128, 0, stream>>>(Qp, Kp, Vtp, Ow);

  gemm128<1, 0><<<ggrid, 256, 0, stream>>>(Ow, Wo, bo, out, MROWS, DD, DD);
}
